// MaxGraphPool_40097814676120
// MI455X (gfx1250) — matrix operand order repaired and checked by reading, not compiled, not run on hardware
//
#include <hip/hip_runtime.h>

typedef __attribute__((ext_vector_type(16))) _Float16 v16h;
typedef __attribute__((ext_vector_type(8)))  float    v8f;
typedef int v4i_gcc __attribute__((__vector_size__(4 * sizeof(int))));

#define AS_GLOBAL __attribute__((address_space(1)))
#define AS_LDS    __attribute__((address_space(3)))

#define B_SZ    32
#define N_SZ    512
#define IN_DIM  64
#define OUT_DIM 64
#define OC      16          // output columns handled per block
#define TILE_N  128         // rows staged per iteration
#define NT      (N_SZ / TILE_N)
#define THREADS 256         // 8 waves (wave32)

__device__ __forceinline__ void wait_asynccnt0() {
#if __has_builtin(__builtin_amdgcn_s_wait_asynccnt)
    __builtin_amdgcn_s_wait_asynccnt(0);
#else
    asm volatile("s_wait_asynccnt 0" ::: "memory");
#endif
}

__global__ __launch_bounds__(THREADS)
void gated_pool_kernel(const float* __restrict__ x,
                       const float* __restrict__ W,
                       const float* __restrict__ bias,
                       float* __restrict__ out)
{
    __shared__ float xs[TILE_N * IN_DIM];   // 32 KB staged x tile (f32)
    __shared__ float gt[TILE_N * OC];       //  8 KB gate tile (f32)

    const int b    = blockIdx.y;
    const int oc   = blockIdx.x;            // o-chunk [oc*16, oc*16+16)
    const int tid  = threadIdx.x;
    const int lane = tid & 31;
    const int w    = tid >> 5;              // wave id 0..7
    const int half = lane >> 4;
    const int l16  = lane & 15;

    // ---- resident B fragments: B[k][o] = W[o][k] for this o-chunk ----
    // B 32x16 (KxN) as loaded here: lane={half,col}; element e -> K = half*16 + e (contiguous, same as A)
    const int o_col = oc * OC + l16;
    v16h bf0, bf1;
    #pragma unroll
    for (int e = 0; e < 16; ++e) {
        bf0[e] = (_Float16)W[o_col * IN_DIM +      (half * 16 + e)];
        bf1[e] = (_Float16)W[o_col * IN_DIM + 32 + (half * 16 + e)];
    }
    const float bias_o = bias[o_col];

    // acc[i][o'] with i = 2*lane, 2*lane+1 ; o' = w*2 + {0,1}
    float acc00 = -INFINITY, acc01 = -INFINITY;
    float acc10 = -INFINITY, acc11 = -INFINITY;

    const float* xb = x + (size_t)b * N_SZ * IN_DIM;

    for (int t = 0; t < NT; ++t) {
        __syncthreads();   // protect xs/gt from previous iteration's readers

        // ---- stage x tile (128x64 f32) into LDS ----
        const float4* src = (const float4*)(xb + t * TILE_N * IN_DIM);
        float4*       dst = (float4*)xs;
#if defined(__gfx1250__) && __has_builtin(__builtin_amdgcn_global_load_async_to_lds_b128)
        // CDNA5 async direct global->LDS copy (no VGPR round-trip), ASYNCcnt tracked
        #pragma unroll
        for (int idx = tid; idx < TILE_N * IN_DIM / 4; idx += THREADS) {
            __builtin_amdgcn_global_load_async_to_lds_b128(
                (AS_GLOBAL v4i_gcc*)(src + idx),
                (AS_LDS v4i_gcc*)(dst + idx), 0, 0);
        }
        wait_asynccnt0();
#else
        for (int idx = tid; idx < TILE_N * IN_DIM / 4; idx += THREADS)
            dst[idx] = src[idx];
#endif
        __syncthreads();

        // ---- GEMM tile: rows [w*16, w*16+16) x this o-chunk, WMMA f16 ----
        // A 16x32 (MxK) as loaded here: lane={half,m}; element e -> K = half*16 + e (contiguous, same as B)
        v16h a0, a1;
        const int row = w * 16 + l16;
        #pragma unroll
        for (int e = 0; e < 16; ++e) {
            const int k = (half << 4) + e;
            a0[e] = (_Float16)xs[row * IN_DIM + k];
            a1[e] = (_Float16)xs[row * IN_DIM + 32 + k];
        }
        v8f c = {};
        c = __builtin_amdgcn_wmma_f32_16x16x32_f16(false, a0, false, bf0, (short)0, c, false, false);
        c = __builtin_amdgcn_wmma_f32_16x16x32_f16(false, a1, false, bf1, (short)0, c, false, false);

        // sigmoid(c + bias) -> gate tile in LDS (v_rcp_f32 fast path, not IEEE div)
        // C layout: element e -> row m = e + 8*half, col = l16
        #pragma unroll
        for (int e = 0; e < 8; ++e) {
            const float tv = c[e] + bias_o;
            const float g  = __builtin_amdgcn_rcpf(1.0f + __expf(-tv));
            gt[(w * 16 + e + half * 8) * OC + l16] = g;
        }
        __syncthreads();

        // ---- gated outer-product running max over this tile's rows ----
        // lane owns i = {2*lane, 2*lane+1}; wave owns o' = {w*2, w*2+1}
        const float2* xs2 = (const float2*)xs;
        const float2* gt2 = (const float2*)gt;
        #pragma unroll 4
        for (int n = 0; n < TILE_N; ++n) {
            const float2 xv = xs2[n * (IN_DIM / 2) + lane];   // ds_load_b64
            const float2 gv = gt2[n * (OC / 2) + w];          // ds_load_b64 broadcast
            acc00 = fmaxf(acc00, xv.x * gv.x);
            acc01 = fmaxf(acc01, xv.x * gv.y);
            acc10 = fmaxf(acc10, xv.y * gv.x);
            acc11 = fmaxf(acc11, xv.y * gv.y);
        }
    }

    // ---- mean over i = sum of 64 i-values / 64 ----
    float s0 = acc00 + acc10;
    float s1 = acc01 + acc11;
    #pragma unroll
    for (int off = 16; off > 0; off >>= 1) {
        s0 += __shfl_down(s0, off, 32);
        s1 += __shfl_down(s1, off, 32);
    }
    if (lane == 0) {
        out[b * OUT_DIM + oc * OC + w * 2 + 0] = s0 * (1.0f / 64.0f);
        out[b * OUT_DIM + oc * OC + w * 2 + 1] = s1 * (1.0f / 64.0f);
    }
}

extern "C" void kernel_launch(void* const* d_in, const int* in_sizes, int n_in,
                              void* d_out, int out_size, void* d_ws, size_t ws_size,
                              hipStream_t stream) {
    (void)in_sizes; (void)n_in; (void)d_ws; (void)ws_size; (void)out_size;
    const float* x  = (const float*)d_in[0];   // (32, 512, 64) f32
    const float* W  = (const float*)d_in[1];   // (64, 64) f32
    const float* bi = (const float*)d_in[2];   // (64,) f32
    float* out = (float*)d_out;                // (32, 64) f32

    dim3 grid(OUT_DIM / OC, B_SZ);             // (4, 32) = 128 workgroups
    gated_pool_kernel<<<grid, THREADS, 0, stream>>>(x, W, bi, out);
}
